// MaxRetrievalModel_90984587199062
// MI455X (gfx1250) — compile-verified, matrix operand order checked by reading, not run on hardware
//
#include <hip/hip_runtime.h>
#include <hip/hip_bf16.h>
#include <math.h>

typedef __bf16 bf16;
typedef __attribute__((ext_vector_type(16))) __bf16 v16bf;
typedef __attribute__((ext_vector_type(8)))  float  v8f;
typedef __attribute__((ext_vector_type(4)))  float  v4f;

#define D_EMB     128
#define N_ITEMS   16384
#define BATCH     32
#define ITEM_DIM  64
#define N_CLASSES 10

// workspace byte offsets
#define WFRAG_OFF   0u          // 112 frags * 512 bf16 = 114688 B
#define QBUF_OFF    131072u     // 32*128 f32
#define ZBUF_OFF    147456u     // 32*128 f32
#define SCORES_OFF  163840u     // 32*16384 f32 = 2 MB
#define VBUF_OFF    2260992u    // 32*16384*128 bf16 = 128 MB

// fragment-table bases (units of one 512-bf16 fragment)
#define FB_WX1 0
#define FB_WX2 16
#define FB_WKP 48
#define FB_WVP 80

__device__ __forceinline__ bf16 f2bf(float f) {
  unsigned u = __builtin_bit_cast(unsigned, f);
  unsigned r = (u + 0x7FFFu + ((u >> 16) & 1u)) >> 16;
  unsigned short s = (unsigned short)r;
  return __builtin_bit_cast(bf16, s);
}
__device__ __forceinline__ float bfbits2f(unsigned short s) {
  unsigned u = ((unsigned)s) << 16;
  return __builtin_bit_cast(float, u);
}
__device__ __forceinline__ float gelu_exact(float x) {
  return 0.5f * x * (1.0f + erff(x * 0.7071067811865475f));
}

// ---------------------------------------------------------------------------
// Repack fp32 weights [K x 128] into bf16 WMMA B-fragments.
// frag f, lane, elem i:  K = kstep*32 + (lane>>4)*16 + i ; N = ntile*16 + (lane&15)
// storage: wfrag[f*512 + lane*16 + i]
// ---------------------------------------------------------------------------
__global__ void prep_weights(const float* __restrict__ wx1, const float* __restrict__ wx2,
                             const float* __restrict__ wkp, const float* __restrict__ wvp,
                             bf16* __restrict__ wfrag) {
  int f = blockIdx.x;            // 0..111
  int t = threadIdx.x;           // 0..511
  int lane = t >> 4, i = t & 15;
  const float* W; int base, ksteps;
  if      (f < 16) { W = wx1; base = FB_WX1; ksteps = 2; }
  else if (f < 48) { W = wx2; base = FB_WX2; ksteps = 4; }
  else if (f < 80) { W = wkp; base = FB_WKP; ksteps = 4; }
  else             { W = wvp; base = FB_WVP; ksteps = 4; }
  int fm = f - base;
  int ntile = fm / ksteps, kk = fm % ksteps;
  int krow = kk * 32 + (lane >> 4) * 16 + i;
  int col  = ntile * 16 + (lane & 15);
  wfrag[(size_t)f * 512 + t] = f2bf(W[krow * 128 + col]);
}

// ---------------------------------------------------------------------------
// Query path: q[b] = mlp(x_query[b]) @ wqp + bqp     (tiny, VALU)
// ---------------------------------------------------------------------------
__global__ void query_mlp(const float* __restrict__ x_query,
                          const float* __restrict__ wq1, const float* __restrict__ bq1,
                          const float* __restrict__ wq2, const float* __restrict__ bq2,
                          const float* __restrict__ wqp, const float* __restrict__ bqp,
                          float* __restrict__ qbuf) {
  __shared__ float h1[128];
  __shared__ float h2[128];
  int b = blockIdx.x, d = threadIdx.x;
  float s = x_query[b];
  h1[d] = gelu_exact(s * wq1[d] + bq1[d]);
  __syncthreads();
  float acc = bq2[d];
  for (int k = 0; k < 128; ++k) acc += h1[k] * wq2[k * 128 + d];
  h2[d] = acc;
  __syncthreads();
  acc = bqp[d];
  for (int k = 0; k < 128; ++k) acc += h2[k] * wqp[k * 128 + d];
  qbuf[b * 128 + d] = acc;
}

// ---------------------------------------------------------------------------
// Hot kernel: per wave a 16-row item tile, 4 chained WMMA GEMMs.
//   h1 = gelu(x @ wx1 + bx1) ; h = h1 @ wx2 + bx2
//   k  = h @ wkp + bkp  -> scores = (k . q) * d^-1/2
//   v  = h @ wvp + bvp  -> stored bf16 to vbuf
// Activations restaged through wave-private LDS (row-major bf16) so the next
// A fragment is a contiguous 32B per-lane read.
// ---------------------------------------------------------------------------
__launch_bounds__(256)
__global__ void item_pipeline(const float* __restrict__ x_items,
                              const float* __restrict__ bx1,
                              const float* __restrict__ bx2,
                              const float* __restrict__ bkp,
                              const float* __restrict__ bvp,
                              const bf16*  __restrict__ wfrag,
                              const float* __restrict__ qbuf,
                              float* __restrict__ scores,
                              bf16*  __restrict__ vbuf) {
  __shared__ __align__(32) bf16 stage[8 * 16 * 128];   // 32 KB, 4 KB per wave
  const int wave  = threadIdx.x >> 5;
  const int lane  = threadIdx.x & 31;
  const int b     = blockIdx.y;
  const int r0    = blockIdx.x * 128 + wave * 16;
  const int rlo   = lane & 15;
  const int khalf = lane >> 4;
  bf16* my = stage + wave * (16 * 128);

  // ---- A fragments from x_items (16 x 64 fp32 -> 2 bf16 frags) ----
  v16bf a0, a1;
  {
    const float* xr = x_items + ((size_t)b * N_ITEMS + r0 + rlo) * ITEM_DIM + khalf * 16;
    v4f p0 = *(const v4f*)(xr + 0);
    v4f p1 = *(const v4f*)(xr + 4);
    v4f p2 = *(const v4f*)(xr + 8);
    v4f p3 = *(const v4f*)(xr + 12);
#pragma unroll
    for (int i = 0; i < 4; ++i) {
      a0[i] = f2bf(p0[i]); a0[4 + i] = f2bf(p1[i]);
      a0[8 + i] = f2bf(p2[i]); a0[12 + i] = f2bf(p3[i]);
    }
    const float* xs = xr + 32;
    p0 = *(const v4f*)(xs + 0); p1 = *(const v4f*)(xs + 4);
    p2 = *(const v4f*)(xs + 8); p3 = *(const v4f*)(xs + 12);
#pragma unroll
    for (int i = 0; i < 4; ++i) {
      a1[i] = f2bf(p0[i]); a1[4 + i] = f2bf(p1[i]);
      a1[8 + i] = f2bf(p2[i]); a1[12 + i] = f2bf(p3[i]);
    }
  }

  v8f acc[8];

  // ---- GEMM1: h1 = x @ wx1 + bx1 (K=64 -> 2 k-steps) ----
#pragma unroll
  for (int t = 0; t < 8; ++t) {
    float bias = bx1[t * 16 + rlo];
    v8f c;
#pragma unroll
    for (int j = 0; j < 8; ++j) c[j] = bias;
    v16bf w0 = *(const v16bf*)(wfrag + (size_t)(FB_WX1 + t * 2 + 0) * 512 + lane * 16);
    c = __builtin_amdgcn_wmma_f32_16x16x32_bf16(false, a0, false, w0, (short)0, c, false, false);
    v16bf w1 = *(const v16bf*)(wfrag + (size_t)(FB_WX1 + t * 2 + 1) * 512 + lane * 16);
    c = __builtin_amdgcn_wmma_f32_16x16x32_bf16(false, a1, false, w1, (short)0, c, false, false);
    acc[t] = c;
  }

  // gelu + stage to LDS (row-major bf16), then reload as A fragments
#pragma unroll
  for (int t = 0; t < 8; ++t)
#pragma unroll
    for (int j = 0; j < 8; ++j)
      my[(j + 8 * khalf) * 128 + t * 16 + rlo] = f2bf(gelu_exact(acc[t][j]));
  asm volatile("s_wait_dscnt 0x0" ::: "memory");
  __builtin_amdgcn_wave_barrier();

  v16bf h0, h1, h2, h3;
  {
    const bf16* ar = my + rlo * 128 + khalf * 16;
    h0 = *(const v16bf*)(ar + 0);
    h1 = *(const v16bf*)(ar + 32);
    h2 = *(const v16bf*)(ar + 64);
    h3 = *(const v16bf*)(ar + 96);
  }
  asm volatile("" ::: "memory");
  __builtin_amdgcn_wave_barrier();

  // ---- GEMM2: h = h1 @ wx2 + bx2 (K=128 -> 4 k-steps) ----
  auto gemm128 = [&](v16bf A0, v16bf A1, v16bf A2, v16bf A3, int fbase,
                     const float* __restrict__ bias) {
#pragma unroll
    for (int t = 0; t < 8; ++t) {
      float bv = bias[t * 16 + rlo];
      v8f c;
#pragma unroll
      for (int j = 0; j < 8; ++j) c[j] = bv;
      v16bf w0 = *(const v16bf*)(wfrag + (size_t)(fbase + t * 4 + 0) * 512 + lane * 16);
      c = __builtin_amdgcn_wmma_f32_16x16x32_bf16(false, A0, false, w0, (short)0, c, false, false);
      v16bf w1 = *(const v16bf*)(wfrag + (size_t)(fbase + t * 4 + 1) * 512 + lane * 16);
      c = __builtin_amdgcn_wmma_f32_16x16x32_bf16(false, A1, false, w1, (short)0, c, false, false);
      v16bf w2 = *(const v16bf*)(wfrag + (size_t)(fbase + t * 4 + 2) * 512 + lane * 16);
      c = __builtin_amdgcn_wmma_f32_16x16x32_bf16(false, A2, false, w2, (short)0, c, false, false);
      v16bf w3 = *(const v16bf*)(wfrag + (size_t)(fbase + t * 4 + 3) * 512 + lane * 16);
      c = __builtin_amdgcn_wmma_f32_16x16x32_bf16(false, A3, false, w3, (short)0, c, false, false);
      acc[t] = c;
    }
  };

  gemm128(h0, h1, h2, h3, FB_WX2, bx2);

  // stage h (no activation), reload as A fragments
#pragma unroll
  for (int t = 0; t < 8; ++t)
#pragma unroll
    for (int j = 0; j < 8; ++j)
      my[(j + 8 * khalf) * 128 + t * 16 + rlo] = f2bf(acc[t][j]);
  asm volatile("s_wait_dscnt 0x0" ::: "memory");
  __builtin_amdgcn_wave_barrier();

  v16bf g0, g1, g2, g3;
  {
    const bf16* ar = my + rlo * 128 + khalf * 16;
    g0 = *(const v16bf*)(ar + 0);
    g1 = *(const v16bf*)(ar + 32);
    g2 = *(const v16bf*)(ar + 64);
    g3 = *(const v16bf*)(ar + 96);
  }
  asm volatile("" ::: "memory");
  __builtin_amdgcn_wave_barrier();

  // ---- GEMM3: k = h @ wkp + bkp ; scores = (k . q) * scale ----
  gemm128(g0, g1, g2, g3, FB_WKP, bkp);
  {
    float qv[8];
#pragma unroll
    for (int t = 0; t < 8; ++t) qv[t] = qbuf[b * 128 + t * 16 + rlo];
    float p[8];
#pragma unroll
    for (int j = 0; j < 8; ++j) {
      float s = 0.f;
#pragma unroll
      for (int t = 0; t < 8; ++t) s += acc[t][j] * qv[t];
      p[j] = s;
    }
#pragma unroll
    for (int m = 1; m < 16; m <<= 1)
#pragma unroll
      for (int j = 0; j < 8; ++j) p[j] += __shfl_xor(p[j], m, 16);
    if (rlo == 0) {
      const float scale = 0.08838834764831845f;   // 128^-0.5
#pragma unroll
      for (int j = 0; j < 8; ++j)
        scores[(size_t)b * N_ITEMS + r0 + khalf * 8 + j] = p[j] * scale;
    }
  }

  // ---- GEMM4: v = h @ wvp + bvp ; bf16 to vbuf via LDS restage ----
  gemm128(g0, g1, g2, g3, FB_WVP, bvp);
#pragma unroll
  for (int t = 0; t < 8; ++t)
#pragma unroll
    for (int j = 0; j < 8; ++j)
      my[(j + 8 * khalf) * 128 + t * 16 + rlo] = f2bf(acc[t][j]);
  asm volatile("s_wait_dscnt 0x0" ::: "memory");
  __builtin_amdgcn_wave_barrier();
  {
    int row = lane >> 1;
    int c0  = (lane & 1) * 64;                    // 64 bf16 = 128 B per lane
    const v4f* src = (const v4f*)(my + row * 128 + c0);
    v4f* dst = (v4f*)(vbuf + ((size_t)b * N_ITEMS + r0 + row) * 128 + c0);
#pragma unroll
    for (int i = 0; i < 8; ++i) dst[i] = src[i];
  }
}

// ---------------------------------------------------------------------------
// Sparsemax (bisection for tau) + z = attn . V, one block per batch.
// ---------------------------------------------------------------------------
__launch_bounds__(256)
__global__ void sparsemax_z(const float* __restrict__ scores,
                            const bf16* __restrict__ vbuf,
                            float* __restrict__ zbuf) {
  __shared__ bf16  attn[N_ITEMS];   // 32 KB
  __shared__ float red[256];
  __shared__ float zred[4][128];
  int b = blockIdx.x, tid = threadIdx.x;
  const float* sb = scores + (size_t)b * N_ITEMS;
  float sv[64];
#pragma unroll
  for (int i = 0; i < 64; ++i) sv[i] = sb[i * 256 + tid];

  float m = -1e30f;
#pragma unroll
  for (int i = 0; i < 64; ++i) m = fmaxf(m, sv[i]);
  red[tid] = m; __syncthreads();
  for (int s = 128; s > 0; s >>= 1) {
    if (tid < s) red[tid] = fmaxf(red[tid], red[tid + s]);
    __syncthreads();
  }
  float zmax = red[0]; __syncthreads();

  float lo = zmax - 1.0f, hi = zmax;
  for (int it = 0; it < 40; ++it) {
    float tau = 0.5f * (lo + hi);
    float ssum = 0.f;
#pragma unroll
    for (int i = 0; i < 64; ++i) ssum += fmaxf(sv[i] - tau, 0.f);
    red[tid] = ssum; __syncthreads();
    for (int s = 128; s > 0; s >>= 1) {
      if (tid < s) red[tid] += red[tid + s];
      __syncthreads();
    }
    float tot = red[0]; __syncthreads();
    if (tot > 1.0f) lo = tau; else hi = tau;
  }
  float tau = 0.5f * (lo + hi);
#pragma unroll
  for (int i = 0; i < 64; ++i) attn[i * 256 + tid] = f2bf(fmaxf(sv[i] - tau, 0.f));
  __syncthreads();

  // z[d] = sum_n attn[n] * v[b][n][d]; thread handles a d-pair, n strided by 4
  int pair = tid & 63;
  int g    = tid >> 6;
  float z0 = 0.f, z1 = 0.f;
  const unsigned* vrow = (const unsigned*)(vbuf + (size_t)b * N_ITEMS * 128) + pair;
  for (int n = g; n < N_ITEMS; n += 4) {
    float w = bfbits2f(__builtin_bit_cast(unsigned short, attn[n]));
    if (w != 0.f) {
      unsigned d2 = vrow[(size_t)n * 64];
      z0 += w * bfbits2f((unsigned short)(d2 & 0xFFFFu));
      z1 += w * bfbits2f((unsigned short)(d2 >> 16));
    }
  }
  zred[g][2 * pair]     = z0;
  zred[g][2 * pair + 1] = z1;
  __syncthreads();
  if (tid < 128)
    zbuf[b * 128 + tid] = zred[0][tid] + zred[1][tid] + zred[2][tid] + zred[3][tid];
}

// ---------------------------------------------------------------------------
// Head MLP: out = gelu(z @ wp1 + bp1) @ wp2 + bp2
// ---------------------------------------------------------------------------
__global__ void final_mlp(const float* __restrict__ zbuf,
                          const float* __restrict__ wp1, const float* __restrict__ bp1,
                          const float* __restrict__ wp2, const float* __restrict__ bp2,
                          float* __restrict__ out) {
  __shared__ float zl[128];
  __shared__ float gl[128];
  int b = blockIdx.x, d = threadIdx.x;
  zl[d] = zbuf[b * 128 + d];
  __syncthreads();
  float acc = bp1[d];
  for (int k = 0; k < 128; ++k) acc += zl[k] * wp1[k * 128 + d];
  gl[d] = gelu_exact(acc);
  __syncthreads();
  if (d < N_CLASSES) {
    float o = bp2[d];
    for (int k = 0; k < 128; ++k) o += gl[k] * wp2[k * N_CLASSES + d];
    out[b * N_CLASSES + d] = o;
  }
}

extern "C" void kernel_launch(void* const* d_in, const int* in_sizes, int n_in,
                              void* d_out, int out_size, void* d_ws, size_t ws_size,
                              hipStream_t stream) {
  (void)in_sizes; (void)n_in; (void)out_size; (void)ws_size;
  const float* x_items = (const float*)d_in[0];
  const float* x_query = (const float*)d_in[1];
  const float* wx1 = (const float*)d_in[2];
  const float* bx1 = (const float*)d_in[3];
  const float* wx2 = (const float*)d_in[4];
  const float* bx2 = (const float*)d_in[5];
  const float* wq1 = (const float*)d_in[6];
  const float* bq1 = (const float*)d_in[7];
  const float* wq2 = (const float*)d_in[8];
  const float* bq2 = (const float*)d_in[9];
  const float* wqp = (const float*)d_in[10];
  const float* bqp = (const float*)d_in[11];
  const float* wkp = (const float*)d_in[12];
  const float* bkp = (const float*)d_in[13];
  const float* wvp = (const float*)d_in[14];
  const float* bvp = (const float*)d_in[15];
  const float* wp1 = (const float*)d_in[16];
  const float* bp1 = (const float*)d_in[17];
  const float* wp2 = (const float*)d_in[18];
  const float* bp2 = (const float*)d_in[19];

  char* ws = (char*)d_ws;
  bf16*  wfrag  = (bf16*) (ws + WFRAG_OFF);
  float* qbuf   = (float*)(ws + QBUF_OFF);
  float* zbuf   = (float*)(ws + ZBUF_OFF);
  float* scoreb = (float*)(ws + SCORES_OFF);
  bf16*  vbuf   = (bf16*) (ws + VBUF_OFF);

  prep_weights<<<112, 512, 0, stream>>>(wx1, wx2, wkp, wvp, wfrag);
  query_mlp<<<BATCH, 128, 0, stream>>>(x_query, wq1, bq1, wq2, bq2, wqp, bqp, qbuf);
  dim3 grid(N_ITEMS / 128, BATCH);
  item_pipeline<<<grid, 256, 0, stream>>>(x_items, bx1, bx2, bkp, bvp, wfrag, qbuf,
                                          scoreb, vbuf);
  sparsemax_z<<<BATCH, 256, 0, stream>>>(scoreb, vbuf, zbuf);
  final_mlp<<<BATCH, 128, 0, stream>>>(zbuf, wp1, bp1, wp2, bp2, (float*)d_out);
}
